// HopfieldAttention_17042430231088
// MI455X (gfx1250) — matrix operand order repaired and checked by reading, not compiled, not run on hardware
//
#include <hip/hip_runtime.h>
#include <hip/hip_bf16.h>

// ---------------------------------------------------------------------------
// Problem constants (match the JAX reference) -- all GEMMs are 4096x1024x1024
// ---------------------------------------------------------------------------
#define BATCH   4
#define SEQ     1024
#define DIMX    1024
#define HEADS   16
#define HD      64
#define INNERD  1024
#define GM      (BATCH*SEQ)           // 4096  (GEMM M)
#define GN      1024                  // GEMM N
#define GK      1024                  // GEMM K
#define NX      (BATCH*SEQ*DIMX)      // 4194304
#define NW      (DIMX*INNERD)         // 1048576
#define LN_EPS  1e-5f
#define STEP_EPS 1e-4f

typedef __attribute__((ext_vector_type(16))) __bf16 bf16x16;
typedef __attribute__((ext_vector_type(8)))  float  f32x8;
typedef __attribute__((ext_vector_type(4)))  unsigned int u32x4;
typedef __attribute__((ext_vector_type(8)))  int  i32x8;
typedef __attribute__((ext_vector_type(4)))  int  i32x4;

#ifdef __has_builtin
#if __has_builtin(__builtin_amdgcn_tensor_load_to_lds)
#define USE_TDM 1
#endif
#if __has_builtin(__builtin_amdgcn_s_wait_tensorcnt)
#define HAVE_TENSORCNT 1
#endif
#endif
#ifndef USE_TDM
#define USE_TDM 0
#endif

union AB {
    bf16x16      v;
    unsigned int w[8];
    unsigned short h[16];
};

__device__ __forceinline__ unsigned short f2bf(float f) {
    unsigned int u = __float_as_uint(f);
    unsigned int r = u + 0x7FFFu + ((u >> 16) & 1u);   // round-to-nearest-even
    return (unsigned short)(r >> 16);
}
__device__ __forceinline__ unsigned int pack2(float a, float b) {
    return (unsigned int)f2bf(a) | ((unsigned int)f2bf(b) << 16);
}
// B-fragment (K=32): 16 contiguous bf16 for this lane's (n, half).
__device__ __forceinline__ void ld_frag32(const unsigned short* p, AB& f) {
    const uint4* q = reinterpret_cast<const uint4*>(p);
    uint4 d0 = q[0], d1 = q[1];
    f.w[0]=d0.x; f.w[1]=d0.y; f.w[2]=d0.z; f.w[3]=d0.w;
    f.w[4]=d1.x; f.w[5]=d1.y; f.w[6]=d1.z; f.w[7]=d1.w;
}
// A-fragment (16x32 bf16): lane m = lane&15, K = 2v + 16*half.
__device__ __forceinline__ void ld_fragA(const unsigned short* rowp, int half, AB& f) {
    #pragma unroll
    for (int v = 0; v < 8; v++) {
        int klo = 2 * v + 16 * half;
        f.w[v] = *reinterpret_cast<const unsigned int*>(rowp + klo);
    }
}
__device__ __forceinline__ void wait_tensor0() {
#if defined(HAVE_TENSORCNT)
    __builtin_amdgcn_s_wait_tensorcnt(0);
#else
    asm volatile("s_wait_tensorcnt 0x0" ::: "memory");
#endif
}
// Wave-local LDS ordering point: same-wave DS ops are in-order (DScnt);
// the wait + memory clobber pins the store->load handoff without a barrier.
__device__ __forceinline__ void lds_wave_wait() {
    asm volatile("s_wait_dscnt 0x0" ::: "memory");
}

// ---------------------------------------------------------------------------
__global__ void cvt_bf16(const float* __restrict__ in, unsigned short* __restrict__ out, int n) {
    int i = blockIdx.x * 256 + threadIdx.x;
    if (i < n) out[i] = f2bf(in[i]);
}

// ---------------------------------------------------------------------------
// C = A(4096x1024 bf16 row) @ B(1024x1024 bf16 row) + bias ; C fp32 row-major.
// Block 256 thr (8 waves), tile 128x128x32; wave = 32x64 -> 8 WMMA / K-step.
// A tile staged via Tensor Data Mover (TDM) into padded LDS (row stride 40 el);
// B tile staged transposed by all threads into a disjoint LDS region.
// ---------------------------------------------------------------------------
__launch_bounds__(256)
__global__ void gemm_bf16(const unsigned short* __restrict__ A,
                          const unsigned short* __restrict__ Bm,
                          const float* __restrict__ bias,
                          float* __restrict__ C) {
    __shared__ __align__(16) unsigned short As[128 * 40];
    __shared__ __align__(16) unsigned short Bs[128 * 40];   // transposed [n][k]
    const int t    = threadIdx.x;
    const int lane = t & 31;
    const int wave = t >> 5;
    const int wm   = wave >> 1;        // 0..3  (M direction)
    const int wn   = wave & 1;         // 0..1  (N direction)
    const int ln16 = lane & 15;
    const int half = lane >> 4;
    const int nBase = blockIdx.x * 128;
    const int mBase = blockIdx.y * 128;

#if USE_TDM
    // D# group1 for: 2-byte elems, tile 32(x) x 128(rows), tensor row stride GK,
    // LDS padding: +4 DWORDs after every 16 DWORDs (row 64B -> stride 80B).
    const int g1w0 = (1 << 16)            // data_size = 1 (2 bytes)
                   | (1 << 20)            // pad_enable
                   | (3 << 22)            // pad_interval code 3 -> 16 DWORDs
                   | (3 << 25);           // pad_amount  code 3 -> 4 DWORDs
    const i32x8 g1 = { g1w0,
                       (int)((GK & 0xFFFF) << 16),                    // tensor_dim0 lo
                       (int)((GK >> 16) | ((GM & 0xFFFF) << 16)),     // td0 hi | td1 lo
                       (int)((GM >> 16) | (32 << 16)),                // td1 hi | tile_dim0=32
                       (int)128,                                      // tile_dim1=128, tile_dim2=0
                       (int)GK,                                       // tensor_dim0_stride lo
                       0, 0 };
    const i32x4 gz = { 0, 0, 0, 0 };
    const unsigned int ldsA = (unsigned int)(size_t)(void*)As;
#endif

    f32x8 acc[2][4] = {};

    for (int k0 = 0; k0 < GK; k0 += 32) {
#if USE_TDM
        if (wave == 0) {   // one wave drives the DMA of the A tile
            unsigned long long ga =
                (unsigned long long)(size_t)(A + (size_t)mBase * GK + k0);
            u32x4 g0 = { 1u,                                  // count=1, user D#
                         ldsA,
                         (unsigned int)ga,
                         ((unsigned int)(ga >> 32) & 0x01FFFFFFu) | 0x80000000u }; // type=2
#if __clang_major__ >= 23
            const i32x8 gz8 = { 0,0,0,0,0,0,0,0 };
            __builtin_amdgcn_tensor_load_to_lds(g0, g1, gz, gz, gz8, 0);
#else
            __builtin_amdgcn_tensor_load_to_lds(g0, g1, gz, gz, 0);
#endif
        }
#else
        {   // fallback: stage A tile 128x32 manually
            int row  = t >> 1;
            int cseg = (t & 1) << 4;
            const uint4* src = reinterpret_cast<const uint4*>(A + (size_t)(mBase + row) * GK + k0 + cseg);
            uint4 d0 = src[0], d1 = src[1];
            *reinterpret_cast<uint4*>(&As[row * 40 + cseg])     = d0;
            *reinterpret_cast<uint4*>(&As[row * 40 + cseg + 8]) = d1;
        }
#endif
        {   // stage B tile 32x128, transposed into Bs[n][k]
            int kk   = t >> 3;
            int nseg = (t & 7) << 4;
            const unsigned short* gsrc = Bm + (size_t)(k0 + kk) * GN + nBase + nseg;
            const uint4* src = reinterpret_cast<const uint4*>(gsrc);
            uint4 d0 = src[0], d1 = src[1];
            // warm GL2 32 columns further along this B row (global_prefetch_b8)
            if (k0 + 32 < GK) __builtin_prefetch(gsrc + 32, 0, 1);
            const unsigned short* hs = reinterpret_cast<const unsigned short*>(&d0);
            #pragma unroll
            for (int i = 0; i < 8; i++) Bs[(nseg + i) * 40 + kk] = hs[i];
            hs = reinterpret_cast<const unsigned short*>(&d1);
            #pragma unroll
            for (int i = 0; i < 8; i++) Bs[(nseg + 8 + i) * 40 + kk] = hs[i];
        }
#if USE_TDM
        if (wave == 0) wait_tensor0();
#endif
        __syncthreads();

        AB a[2], b[4];
        #pragma unroll
        for (int mt = 0; mt < 2; mt++) {
            int m = wm * 32 + mt * 16 + ln16;
            ld_fragA(&As[m * 40], half, a[mt]);
        }
        #pragma unroll
        for (int nt = 0; nt < 4; nt++) {
            int n = wn * 64 + nt * 16 + ln16;
            ld_frag32(&Bs[n * 40 + 16 * half], b[nt]);
        }
        #pragma unroll
        for (int mt = 0; mt < 2; mt++)
            #pragma unroll
            for (int nt = 0; nt < 4; nt++)
                acc[mt][nt] = __builtin_amdgcn_wmma_f32_16x16x32_bf16(
                    false, a[mt].v, false, b[nt].v, (short)0, acc[mt][nt], false, false);
        __syncthreads();
    }

    // Epilogue: constant row stride (GN*4 bytes) -> base + immediate offsets.
    #pragma unroll
    for (int mt = 0; mt < 2; mt++) {
        #pragma unroll
        for (int nt = 0; nt < 4; nt++) {
            int col = nBase + wn * 64 + nt * 16 + ln16;
            float bv = bias[col];
            float* cp = C + (size_t)(mBase + wm * 32 + mt * 16 + 8 * half) * GN + col;
            #pragma unroll
            for (int r = 0; r < 8; r++)
                cp[r * GN] = acc[mt][nt][r] + bv;
        }
    }
}

// ---------------------------------------------------------------------------
// LayerNorm over head_dim (64) + reorder (B,S,H*D) -> (B,H,S,D).
// ---------------------------------------------------------------------------
__device__ __forceinline__ void ln_row(const float* src, const float* gamma, const float* beta,
                                       int lane, float& o0, float& o1) {
    float v0 = src[lane], v1 = src[lane + 32];
    float s = v0 + v1;
    #pragma unroll
    for (int m = 16; m >= 1; m >>= 1) s += __shfl_xor(s, m, 32);
    float mu = s * (1.f / 64.f);
    float d0 = v0 - mu, d1 = v1 - mu;
    float sq = d0 * d0 + d1 * d1;
    #pragma unroll
    for (int m = 16; m >= 1; m >>= 1) sq += __shfl_xor(sq, m, 32);
    float inv = rsqrtf(sq * (1.f / 64.f) + LN_EPS);
    o0 = d0 * inv * gamma[lane]      + beta[lane];
    o1 = d1 * inv * gamma[lane + 32] + beta[lane + 32];
}

__global__ void ln_reorder_f32(const float* __restrict__ lin, const float* __restrict__ gamma,
                               const float* __restrict__ beta, float* __restrict__ out) {
    int row  = blockIdx.x * 8 + (threadIdx.x >> 5);   // (b*S+s)*H + h
    int lane = threadIdx.x & 31;
    int b = row >> 14, s = (row >> 4) & 1023, h = row & 15;
    const float* src = lin + (size_t)(b * SEQ + s) * INNERD + h * 64;
    float o0, o1;
    ln_row(src, gamma, beta, lane, o0, o1);
    float* dst = out + (size_t)((b * HEADS + h) * SEQ + s) * 64;
    dst[lane] = o0; dst[lane + 32] = o1;
}

__global__ void ln_reorder_bf16(const float* __restrict__ lin, const float* __restrict__ gamma,
                                const float* __restrict__ beta, unsigned short* __restrict__ out) {
    int row  = blockIdx.x * 8 + (threadIdx.x >> 5);
    int lane = threadIdx.x & 31;
    int b = row >> 14, s = (row >> 4) & 1023, h = row & 15;
    const float* src = lin + (size_t)(b * SEQ + s) * INNERD + h * 64;
    float o0, o1;
    ln_row(src, gamma, beta, lane, o0, o1);
    unsigned short* dst = out + (size_t)((b * HEADS + h) * SEQ + s) * 64;
    dst[lane] = f2bf(o0); dst[lane + 32] = f2bf(o1);
}

// v: (B,S,H*D) fp32 -> vT bf16 (B,H,D,S)
__global__ void reorder_vT(const float* __restrict__ lin, unsigned short* __restrict__ vt) {
    int idx = blockIdx.x * 256 + threadIdx.x;
    if (idx >= NX) return;
    int b = idx >> 20, s = (idx >> 10) & 1023, inner = idx & 1023;
    int h = inner >> 6, d = inner & 63;
    vt[((size_t)((b * HEADS + h) * 64 + d) << 10) + s] = f2bf(lin[idx]);
}

// q_final (B,H,S,D) fp32 -> merged bf16 (B,S,H*D)
__global__ void merge_bf16(const float* __restrict__ qf, unsigned short* __restrict__ merged) {
    int idx = blockIdx.x * 256 + threadIdx.x;
    if (idx >= NX) return;
    int b = idx >> 20, s = (idx >> 10) & 1023, inner = idx & 1023;
    int h = inner >> 6, d = inner & 63;
    merged[idx] = f2bf(qf[(size_t)((b * HEADS + h) * SEQ + s) * 64 + d]);
}

// ---------------------------------------------------------------------------
// One Hopfield update step, flash-attention style.
// Wave = one (b,h, 16-row q tile). 32-key tiles: 4 WMMA scores + 4 WMMA PV.
// P (probabilities) lives in wave-private LDS -> wave-local DS waits only,
// no block barriers in the inner loop.
// ---------------------------------------------------------------------------
__launch_bounds__(256)
__global__ void hop_step(const float* __restrict__ qIn, float* __restrict__ qOut,
                         const unsigned short* __restrict__ kh,    // (B,H,S,D) bf16
                         const unsigned short* __restrict__ vt,    // (B,H,D,S) bf16
                         const int* __restrict__ flag, float* __restrict__ sumsq) {
    __shared__ __align__(16) unsigned short P[8 * 16 * 40];
    const int lane = threadIdx.x & 31;
    const int wave = threadIdx.x >> 5;
    const int ln16 = lane & 15;
    const int half = lane >> 4;
    const int task = blockIdx.x * 8 + wave;       // B*H*(S/16) = 4096 tasks
    const int qt   = task & 63;
    const int bh   = task >> 6;
    const float*          qBase = qIn + ((size_t)bh * SEQ + qt * 16) * 64;
    const unsigned short* kBase = kh + (size_t)bh * SEQ * 64;
    const unsigned short* vBase = vt + (size_t)bh * 64 * SEQ;
    unsigned short*       Pw    = P + wave * 16 * 40;
    const int conv = flag[0];

    // q A-fragments (bf16 converted in-register): d in [0,32) and [32,64)
    AB a0, a1;
    #pragma unroll
    for (int v = 0; v < 8; v++) {
        int klo = 2 * v + 16 * half;
        float2 qa = *reinterpret_cast<const float2*>(qBase + ln16 * 64 + klo);
        float2 qb = *reinterpret_cast<const float2*>(qBase + ln16 * 64 + 32 + klo);
        a0.w[v] = pack2(qa.x, qa.y);
        a1.w[v] = pack2(qb.x, qb.y);
    }

    f32x8 o[4] = {};
    float mrow[8], lrow[8];
    #pragma unroll
    for (int r = 0; r < 8; r++) { mrow[r] = -3.0e38f; lrow[r] = 0.f; }
    const float scale = 0.125f;                   // 1/sqrt(64)

    for (int kt = 0; kt < SEQ; kt += 32) {
        f32x8 c[2];
        #pragma unroll
        for (int j2 = 0; j2 < 2; j2++) {
            AB kb0, kb1;
            const unsigned short* kr = kBase + (size_t)(kt + j2 * 16 + ln16) * 64 + 16 * half;
            ld_frag32(kr,      kb0);              // d 0..31
            ld_frag32(kr + 32, kb1);              // d 32..63
            if (kt + 32 < SEQ) __builtin_prefetch(kr + 32 * 64, 0, 1);  // next key tile
            f32x8 z = {};
            z = __builtin_amdgcn_wmma_f32_16x16x32_bf16(false, a0.v, false, kb0.v, (short)0, z, false, false);
            z = __builtin_amdgcn_wmma_f32_16x16x32_bf16(false, a1.v, false, kb1.v, (short)0, z, false, false);
            c[j2] = z;
        }
        lds_wave_wait();   // prior iteration's P fragment loads complete
        #pragma unroll
        for (int r = 0; r < 8; r++) {
            float v0 = c[0][r] * scale, v1 = c[1][r] * scale;
            float mx = fmaxf(v0, v1);
            #pragma unroll
            for (int m = 8; m >= 1; m >>= 1) mx = fmaxf(mx, __shfl_xor(mx, m, 32));
            float mnew = fmaxf(mrow[r], mx);
            float al   = __expf(mrow[r] - mnew);
            float p0   = __expf(v0 - mnew);
            float p1   = __expf(v1 - mnew);
            float ps   = p0 + p1;
            #pragma unroll
            for (int m = 8; m >= 1; m >>= 1) ps += __shfl_xor(ps, m, 32);
            lrow[r] = lrow[r] * al + ps;
            mrow[r] = mnew;
            #pragma unroll
            for (int nt = 0; nt < 4; nt++) o[nt][r] *= al;
            Pw[(r + 8 * half) * 40 + ln16]      = f2bf(p0);
            Pw[(r + 8 * half) * 40 + 16 + ln16] = f2bf(p1);
        }
        lds_wave_wait();   // P stores visible (same-wave DS ops are in-order)
        AB ap;
        ld_fragA(&Pw[ln16 * 40], half, ap);
        #pragma unroll
        for (int nt = 0; nt < 4; nt++) {
            AB vb;
            const unsigned short* vr = vBase + (size_t)(nt * 16 + ln16) * SEQ + kt + 16 * half;
            ld_frag32(vr, vb);
            if (kt + 32 < SEQ) __builtin_prefetch(vr + 32, 0, 1);       // next value tile
            o[nt] = __builtin_amdgcn_wmma_f32_16x16x32_bf16(false, ap.v, false, vb.v, (short)0, o[nt], false, false);
        }
    }

    // Finalize: hoisted base, unrolled immediate offsets (r*64, nt*16 elems).
    float local = 0.f;
    const size_t baseIdx = ((size_t)bh * SEQ + qt * 16 + 8 * half) * 64 + ln16;
    const float* qc_p = qIn  + baseIdx;
    float*       qo_p = qOut + baseIdx;
    #pragma unroll
    for (int r = 0; r < 8; r++) {
        float inv = 1.f / lrow[r];
        #pragma unroll
        for (int nt = 0; nt < 4; nt++) {
            float oval = o[nt][r] * inv;
            float qc = qc_p[r * 64 + nt * 16];
            float df = oval - qc;
            local += df * df;
            qo_p[r * 64 + nt * 16] = conv ? qc : oval;
        }
    }
    atomicAdd(sumsq, local);
}

__global__ void init_scal(int* flag, float* sumsq) {
    if (threadIdx.x == 0) { *flag = 0; *sumsq = 0.f; }
}
__global__ void update_flag(int* flag, float* sumsq) {
    if (threadIdx.x == 0) {
        if (sqrtf(*sumsq) < STEP_EPS) *flag = 1;
        *sumsq = 0.f;
    }
}

// ---------------------------------------------------------------------------
extern "C" void kernel_launch(void* const* d_in, const int* in_sizes, int n_in,
                              void* d_out, int out_size, void* d_ws, size_t ws_size,
                              hipStream_t stream) {
    const float* x     = (const float*)d_in[0];
    const float* Wq    = (const float*)d_in[1];
    const float* bq    = (const float*)d_in[2];
    const float* Wk    = (const float*)d_in[3];
    const float* bk    = (const float*)d_in[4];
    const float* Wv    = (const float*)d_in[5];
    const float* bv    = (const float*)d_in[6];
    const float* Wo    = (const float*)d_in[7];
    const float* bo    = (const float*)d_in[8];
    const float* gamma = (const float*)d_in[9];
    const float* beta  = (const float*)d_in[10];
    float* out = (float*)d_out;

    // Workspace carve-up (~88 MB), 256B aligned regions.
    char* w = (char*)d_ws;
    auto carve = [&](size_t bytes) { void* p = (void*)w; w += (bytes + 255) & ~(size_t)255; return p; };
    unsigned short* x_bf   = (unsigned short*)carve((size_t)NX * 2);
    unsigned short* wq_bf  = (unsigned short*)carve((size_t)NW * 2);
    unsigned short* wk_bf  = (unsigned short*)carve((size_t)NW * 2);
    unsigned short* wv_bf  = (unsigned short*)carve((size_t)NW * 2);
    unsigned short* wo_bf  = (unsigned short*)carve((size_t)NW * 2);
    float*          lin    = (float*)carve((size_t)NX * 4);
    float*          qA     = (float*)carve((size_t)NX * 4);
    float*          qB     = (float*)carve((size_t)NX * 4);
    unsigned short* k_hd   = (unsigned short*)carve((size_t)NX * 2);
    unsigned short* vT     = (unsigned short*)carve((size_t)NX * 2);
    unsigned short* merged = (unsigned short*)carve((size_t)NX * 2);
    float* scal = (float*)carve(256);
    float* sumsq = scal;
    int*   flag  = (int*)(scal + 1);

    const dim3 blk(256);
    const dim3 gX((NX + 255) / 256), gW((NW + 255) / 256);
    const dim3 gGemm(GN / 128, GM / 128);                  // (8, 32)
    const dim3 gLN((BATCH * SEQ * HEADS) / 8);             // 8192 blocks
    const dim3 gHop((BATCH * HEADS * (SEQ / 16)) / 8);     // 512 blocks

    // 0) precision conversion
    cvt_bf16<<<gX, blk, 0, stream>>>(x,  x_bf,  NX);
    cvt_bf16<<<gW, blk, 0, stream>>>(Wq, wq_bf, NW);
    cvt_bf16<<<gW, blk, 0, stream>>>(Wk, wk_bf, NW);
    cvt_bf16<<<gW, blk, 0, stream>>>(Wv, wv_bf, NW);
    cvt_bf16<<<gW, blk, 0, stream>>>(Wo, wo_bf, NW);
    init_scal<<<1, 32, 0, stream>>>(flag, sumsq);

    // 1) q = LN(heads(x@Wq + bq)) -> qA (fp32)
    gemm_bf16<<<gGemm, blk, 0, stream>>>(x_bf, wq_bf, bq, lin);
    ln_reorder_f32<<<gLN, blk, 0, stream>>>(lin, gamma, beta, qA);
    // 2) k = LN(heads(x@Wk + bk)) -> bf16 (B,H,S,D)
    gemm_bf16<<<gGemm, blk, 0, stream>>>(x_bf, wk_bf, bk, lin);
    ln_reorder_bf16<<<gLN, blk, 0, stream>>>(lin, gamma, beta, k_hd);
    // 3) v = heads(x@Wv + bv) -> bf16 transposed (B,H,D,S)
    gemm_bf16<<<gGemm, blk, 0, stream>>>(x_bf, wv_bf, bv, lin);
    reorder_vT<<<gX, blk, 0, stream>>>(lin, vT);

    // 4) 3 Hopfield iterations with convergence gating (ping-pong qA<->qB)
    hop_step<<<gHop, blk, 0, stream>>>(qA, qB, k_hd, vT, flag, sumsq);
    update_flag<<<1, 32, 0, stream>>>(flag, sumsq);
    hop_step<<<gHop, blk, 0, stream>>>(qB, qA, k_hd, vT, flag, sumsq);
    update_flag<<<1, 32, 0, stream>>>(flag, sumsq);
    hop_step<<<gHop, blk, 0, stream>>>(qA, qB, k_hd, vT, flag, sumsq);
    update_flag<<<1, 32, 0, stream>>>(flag, sumsq);

    // 5) merge heads -> bf16, output projection
    merge_bf16<<<gX, blk, 0, stream>>>(qB, merged);
    gemm_bf16<<<gGemm, blk, 0, stream>>>(merged, wo_bf, bo, out);
}
